// PointNetSetAbstraction_11123965297234
// MI455X (gfx1250) — matrix operand order repaired and checked by reading, not compiled, not run on hardware
//
#include <hip/hip_runtime.h>
#include <cfloat>

typedef __attribute__((ext_vector_type(16))) _Float16 v16h;
typedef __attribute__((ext_vector_type(8)))  _Float16 v8h;
typedef __attribute__((ext_vector_type(8)))  float    v8f;

// Pointer types matching the async-LDS builtin signature revealed by the
// compiler diagnostic: int __vector(4) in addrspace(1)/(3).
typedef int v4i_t __attribute__((vector_size(16)));
typedef __attribute__((address_space(1))) v4i_t* g_v4i_ptr;
typedef __attribute__((address_space(3))) v4i_t* l_v4i_ptr;

// Problem constants (match reference)
constexpr int Bx  = 8;
constexpr int Nx  = 4096;
constexpr int CIN = 64;
constexpr int Sx  = 1024;
constexpr int Kx  = 32;
constexpr int Mx  = Bx * Sx * Kx;      // 262144 rows of the feature matrix
constexpr int KD0 = 96;                // 67 padded to 96 (multiple of 32)
constexpr int KD1 = 128;
constexpr int KD2 = 128;
constexpr int O0  = 128;
constexpr int O1  = 128;
constexpr int O2  = 256;
constexpr float EPSx = 1e-5f;

// gfx1250 async global->LDS copy (ASYNCcnt) with portable fallback.
#if defined(__has_builtin)
#  if __has_builtin(__builtin_amdgcn_global_load_async_to_lds_b128)
#    define ASYNC_LDS_COPY 1
#  endif
#  if __has_builtin(__builtin_amdgcn_s_wait_asynccnt)
#    define WAIT_ASYNC0() __builtin_amdgcn_s_wait_asynccnt(0)
#  endif
#endif
#ifndef ASYNC_LDS_COPY
#  define ASYNC_LDS_COPY 0
#endif
#ifndef WAIT_ASYNC0
#  define WAIT_ASYNC0() asm volatile("s_wait_asynccnt 0" ::: "memory")
#endif

// ---------------------------------------------------------------------------
// 1) Farthest point sampling: one block per batch, dist array in LDS,
//    block-wide argmax per step (first-index tie-break like jnp.argmax).
// ---------------------------------------------------------------------------
__global__ __launch_bounds__(256) void fps_kernel(const float* __restrict__ xyz,
                                                  int* __restrict__ fps_idx,
                                                  float* __restrict__ new_xyz) {
    __shared__ float dist[Nx];       // 16 KB
    __shared__ float rv[256];
    __shared__ int   ri[256];
    __shared__ float c3[3];
    __shared__ int   s_cur;

    const int b = blockIdx.x;
    const int t = threadIdx.x;
    const float* x = xyz + (size_t)b * Nx * 3;

    for (int n = t; n < Nx; n += 256) dist[n] = 1e10f;
    if (t == 0) s_cur = 0;
    __syncthreads();

    for (int s = 0; s < Sx; ++s) {
        int cur = s_cur;
        if (t == 0) {
            fps_idx[b * Sx + s] = cur;
            float px = x[cur * 3 + 0], py = x[cur * 3 + 1], pz = x[cur * 3 + 2];
            new_xyz[((size_t)b * Sx + s) * 3 + 0] = px;
            new_xyz[((size_t)b * Sx + s) * 3 + 1] = py;
            new_xyz[((size_t)b * Sx + s) * 3 + 2] = pz;
            c3[0] = px; c3[1] = py; c3[2] = pz;
        }
        __syncthreads();
        const float cx = c3[0], cy = c3[1], cz = c3[2];

        float bestv = -1.0f; int besti = 0x7fffffff;
        for (int n = t; n < Nx; n += 256) {
            float dx = x[n * 3 + 0] - cx;
            float dy = x[n * 3 + 1] - cy;
            float dz = x[n * 3 + 2] - cz;
            float d  = dx * dx + dy * dy + dz * dz;
            float old = dist[n];
            d = (d < old) ? d : old;
            dist[n] = d;
            if (d > bestv || (d == bestv && n < besti)) { bestv = d; besti = n; }
        }
        rv[t] = bestv; ri[t] = besti;
        __syncthreads();
        for (int off = 128; off > 0; off >>= 1) {
            if (t < off) {
                float v2 = rv[t + off]; int i2 = ri[t + off];
                if (v2 > rv[t] || (v2 == rv[t] && i2 < ri[t])) { rv[t] = v2; ri[t] = i2; }
            }
            __syncthreads();
        }
        if (t == 0) s_cur = ri[0];
        __syncthreads();
    }
}

// ---------------------------------------------------------------------------
// 2) KNN: one wave32 per query point; full dist row in LDS; 32 rounds of
//    wave-min extraction (ascending distances, lowest-index ties).
// ---------------------------------------------------------------------------
__global__ __launch_bounds__(64) void knn_kernel(const float* __restrict__ xyz,
                                                 const float* __restrict__ new_xyz,
                                                 int* __restrict__ nn_idx) {
    __shared__ float dist[2][Nx];    // 32 KB
    const int wave = threadIdx.x >> 5;
    const int lane = threadIdx.x & 31;
    const int samp = blockIdx.x * 2 + wave;     // b*S + s
    const int b    = samp / Sx;
    const float* x = xyz + (size_t)b * Nx * 3;

    const float qx = new_xyz[(size_t)samp * 3 + 0];
    const float qy = new_xyz[(size_t)samp * 3 + 1];
    const float qz = new_xyz[(size_t)samp * 3 + 2];

    float* d = dist[wave];
    for (int n = lane; n < Nx; n += 32) {
        float dx = x[n * 3 + 0] - qx;
        float dy = x[n * 3 + 1] - qy;
        float dz = x[n * 3 + 2] - qz;
        d[n] = dx * dx + dy * dy + dz * dz;
    }
    __syncthreads();

    for (int k = 0; k < Kx; ++k) {
        float bv = FLT_MAX; int bi = 0x7fffffff;
        for (int n = lane; n < Nx; n += 32) {
            float v = d[n];
            if (v < bv || (v == bv && n < bi)) { bv = v; bi = n; }
        }
        for (int off = 16; off > 0; off >>= 1) {
            float ov = __shfl_xor(bv, off, 32);
            int   oi = __shfl_xor(bi, off, 32);
            if (ov < bv || (ov == bv && oi < bi)) { bv = ov; bi = oi; }
        }
        if (lane == 0) {
            nn_idx[(size_t)samp * Kx + k] = bi;
            d[bi] = FLT_MAX;
        }
        __syncthreads();
    }
}

// ---------------------------------------------------------------------------
// 3) Gather + concat -> f16 feature matrix (M x KD0), zero-padded channels.
// ---------------------------------------------------------------------------
__global__ __launch_bounds__(256) void gather_kernel(const float* __restrict__ xyz,
                                                     const float* __restrict__ points,
                                                     const float* __restrict__ new_xyz,
                                                     const int* __restrict__ nn_idx,
                                                     _Float16* __restrict__ feat) {
    const int m = blockIdx.x * 256 + threadIdx.x;
    if (m >= Mx) return;
    const int bs = m / Kx;             // b*S + s
    const int b  = bs / Sx;
    const int n  = nn_idx[m];

    _Float16* row = feat + (size_t)m * KD0;
    const float nx = new_xyz[(size_t)bs * 3 + 0];
    const float ny = new_xyz[(size_t)bs * 3 + 1];
    const float nz = new_xyz[(size_t)bs * 3 + 2];
    const float* p = xyz + ((size_t)b * Nx + n) * 3;
    row[0] = (_Float16)(p[0] - nx);
    row[1] = (_Float16)(p[1] - ny);
    row[2] = (_Float16)(p[2] - nz);
    const float* pt = points + ((size_t)b * Nx + n) * CIN;
    #pragma unroll 4
    for (int c = 0; c < CIN; ++c) row[3 + c] = (_Float16)pt[c];
    for (int c = 3 + CIN; c < KD0; ++c) row[c] = (_Float16)0.0f;
}

// ---------------------------------------------------------------------------
// 4) Weight conversion: f32 (O x C) -> f16 (O x KD) zero-padded.
// ---------------------------------------------------------------------------
__global__ void wconv_kernel(const float* __restrict__ W, _Float16* __restrict__ Wh,
                             int O, int C, int KD) {
    const int i = blockIdx.x * 256 + threadIdx.x;
    if (i >= O * KD) return;
    const int o = i / KD, c = i % KD;
    Wh[i] = (c < C) ? (_Float16)W[o * C + c] : (_Float16)0.0f;
}

__global__ void zero_kernel(float* p, int n) {
    const int i = blockIdx.x * 256 + threadIdx.x;
    if (i < n) p[i] = 0.0f;
}

// ---------------------------------------------------------------------------
// 5) WMMA GEMM: Y[M x O] = A[M x KD](f16) * Wh[O x KD]^T(f16) + bias, f32 acc.
//    blockIdx.y selects a 128-column weight stripe which the whole block
//    stages into LDS once (async global->LDS on gfx1250, ASYNCcnt-tracked),
//    then 8 waves each compute a 16-row x 128-col stripe: B fragments come
//    from ds_load, A fragments stream via global_load_b128 clauses.
//    Fragment layouts (wave32): A/B use contiguous k = 16*lh + s; D per CDNA5 ISA 7.12.2.
// ---------------------------------------------------------------------------
template <int KD, int O>
__global__ __launch_bounds__(256) void gemm_kernel(const _Float16* __restrict__ A,
                                                   const _Float16* __restrict__ Wh,
                                                   const float* __restrict__ bias,
                                                   float* __restrict__ Y) {
    __shared__ _Float16 wlds[128 * KD];          // 24 KB (KD=96) / 32 KB (KD=128)

    const int wid  = threadIdx.x >> 5;
    const int lane = threadIdx.x & 31;
    const int m0   = (blockIdx.x * 8 + wid) * 16;
    const int n0   = blockIdx.y * 128;
    const int lr   = lane & 15;        // row (A) / column (B,D)
    const int lh   = lane >> 4;        // k-half selector

    // ---- Stage the 128 x KD weight stripe into LDS (all 256 threads) ----
    const _Float16* wsrc = Wh + (size_t)n0 * KD;
    constexpr int CHUNKS = (128 * KD) / 8;       // 16-byte chunks
#if ASYNC_LDS_COPY
    for (int c = threadIdx.x; c < CHUNKS; c += 256) {
        const int e = c * 8;
        __builtin_amdgcn_global_load_async_to_lds_b128(
            (g_v4i_ptr)(uintptr_t)(wsrc + e),
            (l_v4i_ptr)(unsigned int)(uintptr_t)&wlds[e],
            0, 0);
    }
    WAIT_ASYNC0();
#else
    for (int c = threadIdx.x; c < CHUNKS; c += 256) {
        const int e = c * 8;
        *(v8h*)&wlds[e] = *(const v8h*)(wsrc + e);
    }
#endif
    __syncthreads();

    v8f acc[8];
    #pragma unroll
    for (int i = 0; i < 8; ++i)
        #pragma unroll
        for (int j = 0; j < 8; ++j) acc[i][j] = 0.0f;

    // A: lane holds row m0+lr; halfs 0..15 = K kb+lh*16 .. kb+lh*16+15
    const _Float16* Abase = A + (size_t)(m0 + lr) * KD + lh * 16;

    #pragma unroll
    for (int k0 = 0; k0 < KD; k0 += 32) {
        v8h lo = *(const v8h*)(Abase + k0);
        v8h hi = *(const v8h*)(Abase + k0 + 8);
        v16h a = __builtin_shufflevector(lo, hi, 0, 1, 2, 3, 4, 5, 6, 7,
                                                 8, 9, 10, 11, 12, 13, 14, 15);
        #pragma unroll
        for (int nt = 0; nt < 8; ++nt) {
            // B: lane holds column; 16 contiguous K-halfs starting at k0 + lh*16
            v16h bf = *(const v16h*)(&wlds[(nt * 16 + lr) * KD + k0 + lh * 16]);
            acc[nt] = __builtin_amdgcn_wmma_f32_16x16x32_f16(
                false, a, false, bf, (short)0, acc[nt], false, false);
        }
    }

    // D: VGPR i -> row m0 + i + lh*8, column n0 + nt*16 + lr
    #pragma unroll
    for (int nt = 0; nt < 8; ++nt) {
        const int n = n0 + nt * 16 + lr;
        const float bn = bias[n];
        #pragma unroll
        for (int i = 0; i < 8; ++i) {
            const int row = m0 + i + lh * 8;
            Y[(size_t)row * O + n] = acc[nt][i] + bn;
        }
    }
}

// ---------------------------------------------------------------------------
// 6) BN statistics: per-channel sum / sumsq with strided rows + f32 atomics.
// ---------------------------------------------------------------------------
template <int O>
__global__ __launch_bounds__(256) void stats_kernel(const float* __restrict__ Y,
                                                    float* __restrict__ sum,
                                                    float* __restrict__ sumsq) {
    const int gid = blockIdx.x * 256 + threadIdx.x;
    const int c   = gid % O;
    const int r0  = gid / O;
    const int rs  = (gridDim.x * 256) / O;
    float s = 0.0f, s2 = 0.0f;
    for (int r = r0; r < Mx; r += rs) {
        float v = Y[(size_t)r * O + c];
        s += v; s2 += v * v;
    }
    atomicAdd(&sum[c], s);
    atomicAdd(&sumsq[c], s2);
}

__global__ void finalize_kernel(const float* __restrict__ sum,
                                const float* __restrict__ sumsq,
                                const float* __restrict__ gamma,
                                const float* __restrict__ beta,
                                float* __restrict__ scale,
                                float* __restrict__ shift, int O) {
    const int c = threadIdx.x;
    if (c < O) {
        const float inv = 1.0f / (float)Mx;
        float mean = sum[c] * inv;
        float var  = sumsq[c] * inv - mean * mean;
        float sc   = rsqrtf(var + EPSx) * gamma[c];
        scale[c] = sc;
        shift[c] = beta[c] - mean * sc;
    }
}

// ---------------------------------------------------------------------------
// 7) BN apply + ReLU -> f16 activations for the next layer.
// ---------------------------------------------------------------------------
__global__ __launch_bounds__(256) void apply_kernel(const float* __restrict__ Y,
                                                    const float* __restrict__ scale,
                                                    const float* __restrict__ shift,
                                                    _Float16* __restrict__ act,
                                                    int O, size_t total) {
    const size_t stride = (size_t)gridDim.x * 256;
    for (size_t i = (size_t)blockIdx.x * 256 + threadIdx.x; i < total; i += stride) {
        const int c = (int)(i % (size_t)O);
        float v = Y[i] * scale[c] + shift[c];
        act[i] = (_Float16)(v > 0.0f ? v : 0.0f);
    }
}

// ---------------------------------------------------------------------------
// 8) Final layer: BN + ReLU + max over K + transpose -> out (B, 256, S).
// ---------------------------------------------------------------------------
__global__ __launch_bounds__(256) void maxout_kernel(const float* __restrict__ Y2,
                                                     const float* __restrict__ scale,
                                                     const float* __restrict__ shift,
                                                     float* __restrict__ out) {
    const int gid = blockIdx.x * 256 + threadIdx.x;
    if (gid >= Bx * Sx * O2) return;
    const int o  = gid % O2;
    const int bs = gid / O2;
    const float sc = scale[o], sh = shift[o];
    const float* p = Y2 + (size_t)bs * Kx * O2 + o;
    float mx = -FLT_MAX;
    #pragma unroll 4
    for (int k = 0; k < Kx; ++k) {
        float v = p[(size_t)k * O2] * sc + sh;
        mx = fmaxf(mx, v);
    }
    mx = fmaxf(mx, 0.0f);                      // ReLU commutes with max
    const int b = bs / Sx, s = bs % Sx;
    out[((size_t)b * O2 + o) * Sx + s] = mx;
}

// ---------------------------------------------------------------------------
extern "C" void kernel_launch(void* const* d_in, const int* in_sizes, int n_in,
                              void* d_out, int out_size, void* d_ws, size_t ws_size,
                              hipStream_t stream) {
    const float* xyz    = (const float*)d_in[0];
    const float* points = (const float*)d_in[1];
    const float* W0 = (const float*)d_in[2];
    const float* b0 = (const float*)d_in[3];
    const float* g0 = (const float*)d_in[4];
    const float* be0 = (const float*)d_in[5];
    const float* W1 = (const float*)d_in[6];
    const float* b1 = (const float*)d_in[7];
    const float* g1 = (const float*)d_in[8];
    const float* be1 = (const float*)d_in[9];
    const float* W2 = (const float*)d_in[10];
    const float* b2 = (const float*)d_in[11];
    const float* g2 = (const float*)d_in[12];
    const float* be2 = (const float*)d_in[13];

    float* out     = (float*)d_out;
    float* new_xyz = out;                              // (B,S,3) first
    float* outF    = out + (size_t)Bx * Sx * 3;        // (B,256,S) second

    // Workspace carve-out (256B aligned slices).
    char* ws = (char*)d_ws;
    size_t off = 0;
    auto alloc = [&](size_t bytes) -> void* {
        void* p = ws + off;
        off = (off + bytes + 255) & ~(size_t)255;
        return p;
    };
    int*      fps_idx = (int*)alloc((size_t)Bx * Sx * sizeof(int));
    int*      nn_idx  = (int*)alloc((size_t)Bx * Sx * Kx * sizeof(int));
    _Float16* Wh0     = (_Float16*)alloc((size_t)O0 * KD0 * sizeof(_Float16));
    _Float16* Wh1     = (_Float16*)alloc((size_t)O1 * KD1 * sizeof(_Float16));
    _Float16* Wh2     = (_Float16*)alloc((size_t)O2 * KD2 * sizeof(_Float16));
    float*    stsum   = (float*)alloc(256 * sizeof(float));
    float*    stsq    = (float*)alloc(256 * sizeof(float));
    float*    stscale = (float*)alloc(256 * sizeof(float));
    float*    stshift = (float*)alloc(256 * sizeof(float));
    _Float16* feat0   = (_Float16*)alloc((size_t)Mx * KD0 * sizeof(_Float16));
    _Float16* act1    = (_Float16*)alloc((size_t)Mx * KD1 * sizeof(_Float16));
    _Float16* act2    = (_Float16*)alloc((size_t)Mx * KD2 * sizeof(_Float16));
    float*    Ybuf    = (float*)alloc((size_t)Mx * O2 * sizeof(float));

    // Stage 1: sampling / grouping
    fps_kernel<<<Bx, 256, 0, stream>>>(xyz, fps_idx, new_xyz);
    knn_kernel<<<(Bx * Sx) / 2, 64, 0, stream>>>(xyz, new_xyz, nn_idx);
    gather_kernel<<<Mx / 256, 256, 0, stream>>>(xyz, points, new_xyz, nn_idx, feat0);

    // Stage 2: weight conversion to f16 (padded)
    wconv_kernel<<<(O0 * KD0 + 255) / 256, 256, 0, stream>>>(W0, Wh0, O0, 3 + CIN, KD0);
    wconv_kernel<<<(O1 * KD1 + 255) / 256, 256, 0, stream>>>(W1, Wh1, O1, 128, KD1);
    wconv_kernel<<<(O2 * KD2 + 255) / 256, 256, 0, stream>>>(W2, Wh2, O2, 128, KD2);

    // Layer 0: GEMM -> stats -> BN/ReLU -> act1
    zero_kernel<<<2, 256, 0, stream>>>(stsum, 256);
    zero_kernel<<<2, 256, 0, stream>>>(stsq, 256);
    gemm_kernel<KD0, O0><<<dim3((Mx / 16) / 8, O0 / 128), 256, 0, stream>>>(feat0, Wh0, b0, Ybuf);
    stats_kernel<O0><<<256, 256, 0, stream>>>(Ybuf, stsum, stsq);
    finalize_kernel<<<1, 256, 0, stream>>>(stsum, stsq, g0, be0, stscale, stshift, O0);
    apply_kernel<<<8192, 256, 0, stream>>>(Ybuf, stscale, stshift, act1, O0,
                                           (size_t)Mx * O0);

    // Layer 1: GEMM -> stats -> BN/ReLU -> act2
    zero_kernel<<<2, 256, 0, stream>>>(stsum, 256);
    zero_kernel<<<2, 256, 0, stream>>>(stsq, 256);
    gemm_kernel<KD1, O1><<<dim3((Mx / 16) / 8, O1 / 128), 256, 0, stream>>>(act1, Wh1, b1, Ybuf);
    stats_kernel<O1><<<256, 256, 0, stream>>>(Ybuf, stsum, stsq);
    finalize_kernel<<<1, 256, 0, stream>>>(stsum, stsq, g1, be1, stscale, stshift, O1);
    apply_kernel<<<8192, 256, 0, stream>>>(Ybuf, stscale, stshift, act2, O1,
                                           (size_t)Mx * O1);

    // Layer 2: GEMM -> stats -> fused BN/ReLU/max-over-K/transpose
    zero_kernel<<<2, 256, 0, stream>>>(stsum, 256);
    zero_kernel<<<2, 256, 0, stream>>>(stsq, 256);
    gemm_kernel<KD2, O2><<<dim3((Mx / 16) / 8, O2 / 128), 256, 0, stream>>>(act2, Wh2, b2, Ybuf);
    stats_kernel<O2><<<256, 256, 0, stream>>>(Ybuf, stsum, stsq);
    finalize_kernel<<<1, 256, 0, stream>>>(stsum, stsq, g2, be2, stscale, stshift, O2);
    maxout_kernel<<<(Bx * Sx * O2) / 256, 256, 0, stream>>>(Ybuf, stscale, stshift, outF);

    (void)in_sizes; (void)n_in; (void)out_size; (void)ws_size; (void)fps_idx;
}
